// BaseH2XAttLayer_89713276879183
// MI455X (gfx1250) — matrix operand order repaired and checked by reading, not compiled, not run on hardware
//
#include <hip/hip_runtime.h>

typedef __attribute__((ext_vector_type(16))) _Float16 v16h;
typedef __attribute__((ext_vector_type(8)))  float    v8f;

// ---------------------------------------------------------------------------
// WMMA fragment helpers (CDNA5 16x16x32 f16, wave32).
// A (16x32, 16-bit): lane<16 -> M=lane, K = {kb+0..7, kb+16..23}
//                    lane>=16 -> M=lane-16, K = {kb+8..15, kb+24..31}
// B (32x16, 16-bit): lane -> N=lane%16, K = kb + (lane<16 ? 0 : 8) + (i&7) + (i>=8 ? 16 : 0)
// C/D (16x16 f32):   lane -> N=lane%16, VGPR r -> M = r + (lane<16 ? 0 : 8)
// ---------------------------------------------------------------------------
__device__ __forceinline__ v16h load_A_lds(const _Float16* tile, int pitch,
                                           int kbase, int lane) {
  int row = lane & 15;
  int hi  = (lane & 16) ? 8 : 0;
  const _Float16* p0 = tile + row * pitch + kbase + hi;
  const _Float16* p1 = tile + row * pitch + kbase + 16 + hi;
  v16h a;
#pragma unroll
  for (int i = 0; i < 8; ++i) { a[i] = p0[i]; a[8 + i] = p1[i]; }
  return a;
}

__device__ __forceinline__ void store_C(float* out, int pitch, int colbase,
                                        v8f c, float bias, int lane) {
  int n  = colbase + (lane & 15);
  int m0 = (lane & 16) ? 8 : 0;
#pragma unroll
  for (int r = 0; r < 8; ++r) out[(m0 + r) * pitch + n] = c[r] + bias;
}

// Sign-aware float atomic max (init must be -inf).
__device__ __forceinline__ void atomicMaxF(float* addr, float v) {
  if (v >= 0.0f) atomicMax((int*)addr, __float_as_int(v));
  else           atomicMin((unsigned int*)addr, __float_as_uint(v));
}

// ---------------------------------------------------------------------------
// One 16-row MLP tile using fragment-major preswizzled f16 weights.
// Weight fragment layout: frag(slab, ks) at [(slab*NK + ks)*512 + lane*16],
// 16 halfs per lane = one 32B contiguous aligned load.
// 256 threads = 8 waves. GEMM1 always produces 128 hidden cols (8 slabs).
// ---------------------------------------------------------------------------
template <int K1PAD, int OUTC>
__device__ __forceinline__ void mlp_tile(
    const _Float16* s_in,
    const _Float16* __restrict__ W1f, const float* __restrict__ b1,
    const float* __restrict__ lnw, const float* __restrict__ lnb,
    const _Float16* __restrict__ W2f, const float* __restrict__ b2,
    float* s_hid, _Float16* s_hidh, float* s_out, int tid) {
  const int lane = tid & 31;
  const int wave = tid >> 5;
  const int NK1  = K1PAD / 32;

  // GEMM1: [16,K1PAD] x [K1,128] -> s_hid
  {
    const _Float16* wp = W1f + (wave * NK1) * 512 + lane * 16;
    v8f c = {};
#pragma unroll
    for (int ks = 0; ks < NK1; ++ks) {
      v16h a = load_A_lds(s_in, K1PAD, ks * 32, lane);
      v16h b = *(const v16h*)(wp + ks * 512);
      c = __builtin_amdgcn_wmma_f32_16x16x32_f16(false, a, false, b,
                                                 (short)0, c, false, false);
    }
    float bias = b1[wave * 16 + (lane & 15)];
    store_C(s_hid, 128, wave * 16, c, bias, lane);
  }
  __syncthreads();

  // LayerNorm + ReLU: 16 threads per row, 8 cols each; shfl_xor reduce.
  {
    int row = tid >> 4, g = tid & 15;
    const float* hr = s_hid + row * 128 + g * 8;
    float x[8], s = 0.0f, ss = 0.0f;
#pragma unroll
    for (int i = 0; i < 8; ++i) { x[i] = hr[i]; s += x[i]; ss += x[i] * x[i]; }
#pragma unroll
    for (int m = 8; m >= 1; m >>= 1) {
      s  += __shfl_xor(s,  m, 16);
      ss += __shfl_xor(ss, m, 16);
    }
    float mu  = s * (1.0f / 128.0f);
    float var = ss * (1.0f / 128.0f) - mu * mu;
    float inv = rsqrtf(var + 1e-5f);
#pragma unroll
    for (int i = 0; i < 8; ++i) {
      int col = g * 8 + i;
      float y = (x[i] - mu) * inv * lnw[col] + lnb[col];
      s_hidh[row * 128 + col] = (_Float16)fmaxf(y, 0.0f);
    }
  }
  __syncthreads();

  // GEMM2: [16,128] x [128,OUTC] -> s_out  (NK2 = 4)
  if (wave < (OUTC >> 4)) {
    const _Float16* wp = W2f + (wave * 4) * 512 + lane * 16;
    v8f c = {};
#pragma unroll
    for (int ks = 0; ks < 4; ++ks) {
      v16h a = load_A_lds(s_hidh, 128, ks * 32, lane);
      v16h b = *(const v16h*)(wp + ks * 512);
      c = __builtin_amdgcn_wmma_f32_16x16x32_f16(false, a, false, b,
                                                 (short)0, c, false, false);
    }
    float bias = b2[wave * 16 + (lane & 15)];
    store_C(s_out, OUTC, wave * 16, c, bias, lane);
  }
  __syncthreads();
}

// ---------------------------------------------------------------------------
// Kernel P: convert + swizzle all weight matrices into fragment-major f16.
// Region table (units of 512-half fragments):
//   [0,72)    xk_W1  nslab=8 nk=9 Ktot=280 ldw=128
//   [72,104)  xk_W2  8x4, 128, 128
//   [104,176) xv_W1  8x9, 280, 128
//   [176,180) xv_W2  1x4, 128, 16
//   [180,212) xq_W1  8x4, 128, 128
//   [212,244) xq_W2  8x4, 128, 128
// ---------------------------------------------------------------------------
#define NFRAG_TOTAL 244

__global__ void __launch_bounds__(256) k_prep(
    const float* __restrict__ xk_W1, const float* __restrict__ xk_W2,
    const float* __restrict__ xv_W1, const float* __restrict__ xv_W2,
    const float* __restrict__ xq_W1, const float* __restrict__ xq_W2,
    _Float16* __restrict__ wsw) {
  int idx = blockIdx.x * 256 + threadIdx.x;
  int frag = idx >> 9;
  if (frag >= NFRAG_TOTAL) return;
  int within = idx & 511;
  int lane = within >> 4, i = within & 15;

  const float* W; int nk, Ktot, ldw, rbase;
  if      (frag < 72)  { W = xk_W1; nk = 9; Ktot = 280; ldw = 128; rbase = 0;   }
  else if (frag < 104) { W = xk_W2; nk = 4; Ktot = 128; ldw = 128; rbase = 72;  }
  else if (frag < 176) { W = xv_W1; nk = 9; Ktot = 280; ldw = 128; rbase = 104; }
  else if (frag < 180) { W = xv_W2; nk = 4; Ktot = 128; ldw = 16;  rbase = 176; }
  else if (frag < 212) { W = xq_W1; nk = 4; Ktot = 128; ldw = 128; rbase = 180; }
  else                 { W = xq_W2; nk = 4; Ktot = 128; ldw = 128; rbase = 212; }

  int lf = frag - rbase;
  int slab = lf / nk, ks = lf - slab * nk;
  int n = slab * 16 + (lane & 15);
  int k = ks * 32 + ((lane & 16) ? 8 : 0) + (i & 7) + ((i & 8) ? 16 : 0);
  float v = (k < Ktot) ? W[k * ldw + n] : 0.0f;
  wsw[idx] = (_Float16)v;
}

// ---------------------------------------------------------------------------
// Kernel 0: init accumulators
// ---------------------------------------------------------------------------
__global__ void __launch_bounds__(256) k_init(float* segmax, float* denom,
                                              float* numer, int N) {
  int idx = blockIdx.x * 256 + threadIdx.x;
  if (idx < N * 16) { segmax[idx] = -__builtin_inff(); denom[idx] = 0.0f; }
  if (idx < N * 48) { numer[idx] = 0.0f; }
}

// ---------------------------------------------------------------------------
// Kernel 1: q = MLP(h)   [N,128] -> [N,128]
// Full tiles stage the contiguous 8KB input block via GLOBAL_LOAD_ASYNC_TO_LDS.
// ---------------------------------------------------------------------------
__global__ void __launch_bounds__(256) k_q(
    const float* __restrict__ h,
    const _Float16* __restrict__ W1f, const float* __restrict__ b1,
    const float* __restrict__ lnw, const float* __restrict__ lnb,
    const _Float16* __restrict__ W2f, const float* __restrict__ b2,
    float* __restrict__ qout, int N) {
  __shared__ float    s_raw[16 * 128];
  __shared__ _Float16 s_in[16 * 128];
  __shared__ float    s_hid[16 * 128];
  __shared__ _Float16 s_hidh[16 * 128];
  __shared__ float    s_out[16 * 128];
  int tid = threadIdx.x;
  int nb  = blockIdx.x * 16;
  int nn  = N - nb; if (nn > 16) nn = 16;

  if (nn == 16) {
    // Async DMA: 2048 contiguous floats -> LDS, 256 lanes x 2 x b128.
#pragma unroll
    for (int it = 0; it < 2; ++it) {
      int off = tid * 4 + it * 1024;  // floats
      unsigned lds = (unsigned)(uintptr_t)&s_raw[off];
      const float* g = h + (size_t)nb * 128 + off;
      asm volatile("global_load_async_to_lds_b128 %0, %1, off"
                   :: "v"(lds), "v"(g) : "memory");
    }
    asm volatile("s_wait_asynccnt 0x0" ::: "memory");
    __syncthreads();
    for (int idx = tid; idx < 16 * 128; idx += 256)
      s_in[idx] = (_Float16)s_raw[idx];
  } else {
    for (int idx = tid; idx < 16 * 128; idx += 256) {
      int row = idx >> 7;
      float v = (row < nn) ? h[(nb + row) * 128 + (idx & 127)] : 0.0f;
      s_in[idx] = (_Float16)v;
    }
  }
  __syncthreads();

  mlp_tile<128, 128>(s_in, W1f, b1, lnw, lnb, W2f, b2,
                     s_hid, s_hidh, s_out, tid);

  for (int idx = tid; idx < 16 * 128; idx += 256) {
    int row = idx >> 7;
    if (row < nn) qout[(nb + row) * 128 + (idx & 127)] = s_out[idx];
  }
}

// ---------------------------------------------------------------------------
// Kernel 2: per-edge-tile k/v MLPs + scores + gate + atomic segment max
// ---------------------------------------------------------------------------
#define KVP 288  // KV_IN=280 padded to multiple of 32

__global__ void __launch_bounds__(256) k_edge(
    const float* __restrict__ h,
    const float* __restrict__ r_feat,
    const float* __restrict__ edge_feat,
    const int*   __restrict__ ei,
    const float* __restrict__ ew_W, const float* __restrict__ ew_b,
    const _Float16* __restrict__ xk_W1f, const float* __restrict__ xk_b1,
    const float* __restrict__ xk_lnw, const float* __restrict__ xk_lnb,
    const _Float16* __restrict__ xk_W2f, const float* __restrict__ xk_b2,
    const _Float16* __restrict__ xv_W1f, const float* __restrict__ xv_b1,
    const float* __restrict__ xv_lnw, const float* __restrict__ xv_lnb,
    const _Float16* __restrict__ xv_W2f, const float* __restrict__ xv_b2,
    const float* __restrict__ q,
    float* __restrict__ scores_g, float* __restrict__ veff_g,
    float* __restrict__ segmax, int E) {
  __shared__ _Float16 s_in[16 * KVP];
  __shared__ float    s_hid[16 * 128];
  __shared__ _Float16 s_hidh[16 * 128];
  __shared__ float    s_k[16 * 128];
  __shared__ float    s_v[16 * 16];
  __shared__ int      s_dst[16], s_src[16];
  __shared__ float    s_ew[16];

  int tid = threadIdx.x;
  int eb  = blockIdx.x * 16;
  int ne  = E - eb; if (ne > 16) ne = 16;

  if (tid < 16) {
    int e = eb + tid;
    int ok = (tid < ne);
    s_src[tid] = ok ? ei[e]     : 0;
    s_dst[tid] = ok ? ei[E + e] : 0;
  }
  __syncthreads();

  // Gather kv_input = [edge_feat(4) | r_feat(20) | h_dst(128) | h_src(128) | pad(8)]
  for (int idx = tid; idx < 16 * KVP; idx += 256) {
    int row = idx / KVP;
    int col = idx - row * KVP;
    float val = 0.0f;
    if (row < ne) {
      int e = eb + row;
      if      (col < 4)   val = edge_feat[e * 4 + col];
      else if (col < 24)  val = r_feat[e * 20 + (col - 4)];
      else if (col < 152) val = h[s_dst[row] * 128 + (col - 24)];
      else if (col < 280) val = h[s_src[row] * 128 + (col - 152)];
    }
    s_in[idx] = (_Float16)val;
  }
  // Edge gate e_w = sigmoid(r_feat @ ew_W + ew_b)
  if (tid < 16 && tid < ne) {
    int e = eb + tid;
    float acc = ew_b[0];
#pragma unroll
    for (int j = 0; j < 20; ++j) acc += r_feat[e * 20 + j] * ew_W[j];
    s_ew[tid] = 1.0f / (1.0f + __expf(-acc));
  }
  __syncthreads();

  // k = MLP_k(kv_input) -> [16,128]
  mlp_tile<KVP, 128>(s_in, xk_W1f, xk_b1, xk_lnw, xk_lnb, xk_W2f, xk_b2,
                     s_hid, s_hidh, s_k, tid);
  // v = MLP_v(kv_input) -> [16,16]
  mlp_tile<KVP, 16>(s_in, xv_W1f, xv_b1, xv_lnw, xv_lnb, xv_W2f, xv_b2,
                    s_hid, s_hidh, s_v, tid);

  // scores[e][head] = dot(q[dst[e]][head,:], k[e][head,:]) / sqrt(8)
  {
    int e = tid >> 4, hd = tid & 15;
    if (e < ne) {
      int d = s_dst[e];
      const float* qr = q + d * 128 + hd * 8;
      const float* kr = s_k + e * 128 + hd * 8;
      float sc = 0.0f;
#pragma unroll
      for (int i = 0; i < 8; ++i) sc += qr[i] * kr[i];
      sc *= 0.35355339059327373f;  // 1/sqrt(8)
      int gi = (eb + e) * 16 + hd;
      scores_g[gi] = sc;
      veff_g[gi]   = s_v[e * 16 + hd] * s_ew[e];
      atomicMaxF(&segmax[d * 16 + hd], sc);
    }
  }
}

// ---------------------------------------------------------------------------
// Kernel 3: scatter-softmax accumulation
// ---------------------------------------------------------------------------
__global__ void __launch_bounds__(256) k_soft(
    const float* __restrict__ scores_g, const float* __restrict__ veff_g,
    const int* __restrict__ ei, const float* __restrict__ rel_x,
    const float* __restrict__ segmax, float* __restrict__ denom,
    float* __restrict__ numer, int E) {
  int idx = blockIdx.x * 256 + threadIdx.x;
  if (idx >= E * 16) return;
  int e = idx >> 4, hd = idx & 15;
  int d = ei[E + e];
  float ex = __expf(scores_g[idx] - segmax[d * 16 + hd]);
  atomicAdd(&denom[d * 16 + hd], ex);
  float vv = ex * veff_g[idx];
  const float* rx = rel_x + e * 3;
  float* np = numer + (d * 16 + hd) * 3;
  atomicAdd(np + 0, vv * rx[0]);
  atomicAdd(np + 1, vv * rx[1]);
  atomicAdd(np + 2, vv * rx[2]);
}

// ---------------------------------------------------------------------------
// Kernel 4: per-node normalize + head mean
// ---------------------------------------------------------------------------
__global__ void __launch_bounds__(256) k_out(
    const float* __restrict__ denom, const float* __restrict__ numer,
    float* __restrict__ out, int N) {
  int n = blockIdx.x * 256 + threadIdx.x;
  if (n >= N) return;
  float a0 = 0.0f, a1 = 0.0f, a2 = 0.0f;
#pragma unroll
  for (int hd = 0; hd < 16; ++hd) {
    float dn = denom[n * 16 + hd];
    if (dn > 0.0f) {
      float inv = 1.0f / dn;
      const float* np = numer + (n * 16 + hd) * 3;
      a0 += np[0] * inv; a1 += np[1] * inv; a2 += np[2] * inv;
    }
  }
  out[n * 3 + 0] = a0 * (1.0f / 16.0f);
  out[n * 3 + 1] = a1 * (1.0f / 16.0f);
  out[n * 3 + 2] = a2 * (1.0f / 16.0f);
}

// ---------------------------------------------------------------------------
extern "C" void kernel_launch(void* const* d_in, const int* in_sizes, int n_in,
                              void* d_out, int out_size, void* d_ws, size_t ws_size,
                              hipStream_t stream) {
  const float* h         = (const float*)d_in[0];
  const float* rel_x     = (const float*)d_in[1];
  const float* r_feat    = (const float*)d_in[2];
  const float* edge_feat = (const float*)d_in[3];
  const int*   ei        = (const int*)d_in[4];
  const float* ew_W      = (const float*)d_in[5];
  const float* ew_b      = (const float*)d_in[6];
  const float* xk_W1 = (const float*)d_in[7],  *xk_b1 = (const float*)d_in[8];
  const float* xk_lnw = (const float*)d_in[9], *xk_lnb = (const float*)d_in[10];
  const float* xk_W2 = (const float*)d_in[11], *xk_b2 = (const float*)d_in[12];
  const float* xv_W1 = (const float*)d_in[13], *xv_b1 = (const float*)d_in[14];
  const float* xv_lnw = (const float*)d_in[15], *xv_lnb = (const float*)d_in[16];
  const float* xv_W2 = (const float*)d_in[17], *xv_b2 = (const float*)d_in[18];
  const float* xq_W1 = (const float*)d_in[19], *xq_b1 = (const float*)d_in[20];
  const float* xq_lnw = (const float*)d_in[21], *xq_lnb = (const float*)d_in[22];
  const float* xq_W2 = (const float*)d_in[23], *xq_b2 = (const float*)d_in[24];

  const int N = in_sizes[0] / 128;
  const int E = in_sizes[1] / 3;

  // ws layout: [f16 swizzled weights: 244*512 halfs] then float buffers.
  _Float16* wsw = (_Float16*)d_ws;
  const _Float16* xk_W1f = wsw + 0;
  const _Float16* xk_W2f = wsw + 72  * 512;
  const _Float16* xv_W1f = wsw + 104 * 512;
  const _Float16* xv_W2f = wsw + 176 * 512;
  const _Float16* xq_W1f = wsw + 180 * 512;
  const _Float16* xq_W2f = wsw + 212 * 512;

  float* fbase   = (float*)d_ws + (NFRAG_TOTAL * 512 / 2);  // 62464 floats
  float* q       = fbase;                            // N*128
  float* scores  = q + (size_t)N * 128;              // E*16
  float* veff    = scores + (size_t)E * 16;          // E*16
  float* segmax  = veff + (size_t)E * 16;            // N*16
  float* denom   = segmax + (size_t)N * 16;          // N*16
  float* numer   = denom + (size_t)N * 16;           // N*48

  dim3 blk(256);
  k_prep<<<(NFRAG_TOTAL * 512 + 255) / 256, blk, 0, stream>>>(
      xk_W1, xk_W2, xv_W1, xv_W2, xq_W1, xq_W2, wsw);
  k_init<<<(N * 48 + 255) / 256, blk, 0, stream>>>(segmax, denom, numer, N);
  k_q<<<(N + 15) / 16, blk, 0, stream>>>(h, xq_W1f, xq_b1, xq_lnw, xq_lnb,
                                         xq_W2f, xq_b2, q, N);
  k_edge<<<(E + 15) / 16, blk, 0, stream>>>(
      h, r_feat, edge_feat, ei, ew_W, ew_b,
      xk_W1f, xk_b1, xk_lnw, xk_lnb, xk_W2f, xk_b2,
      xv_W1f, xv_b1, xv_lnw, xv_lnb, xv_W2f, xv_b2,
      q, scores, veff, segmax, E);
  {
    long long tot = (long long)E * 16;
    k_soft<<<(unsigned)((tot + 255) / 256), blk, 0, stream>>>(
        scores, veff, ei, rel_x, segmax, denom, numer, E);
  }
  k_out<<<(N + 255) / 256, blk, 0, stream>>>(denom, numer, (float*)d_out, N);
}
